// IBRDilated_38525856645312
// MI455X (gfx1250) — matrix operand order repaired and checked by reading, not compiled, not run on hardware
//
#include <hip/hip_runtime.h>

typedef __attribute__((ext_vector_type(16))) _Float16 v16h;
typedef __attribute__((ext_vector_type(8)))  float    v8f;

#define Bn   8
#define Cn   64
#define Hn   128
#define Wn   128
#define TS   16
#define KK   9
#define XT   20          // tile + 2*dilation halo
#define CPS  401         // xs channel-plane stride in floats (20*20 + 1, odd -> no bank conflicts)
#define ZSTR 258         // zs channel stride in halfs (256 + 2)

__device__ __forceinline__ v8f wmma16(v16h a, v16h b, v8f c) {
    return __builtin_amdgcn_wmma_f32_16x16x32_f16(false, a, false, b, (short)0, c, false, false);
}

// A-fragment (16x32 f16, M x K) from row-major f32 matrix, contiguous k order, same as B:
// lanes 0-15: row M=lane, halfs 0..15 = K 0..15
// lanes 16-31: row M=lane-16, halfs 0..15 = K 16..31
__device__ __forceinline__ v16h load_a_frag(const float* __restrict__ Wm, int ld,
                                            int mBase, int kBase, int lane, int mValid) {
    int row = mBase + (lane & 15);
    bool ok = row < mValid;
    int rr  = ok ? row : (mValid - 1);           // clamp: never read OOB
    int k0  = kBase + ((lane >> 4) << 4);
    const float* p = Wm + rr * ld;
    v16h a;
#pragma unroll
    for (int i = 0; i < 8; ++i) a[i]     = ok ? (_Float16)p[k0 + i]      : (_Float16)0.0f;
#pragma unroll
    for (int i = 0; i < 8; ++i) a[i + 8] = ok ? (_Float16)p[k0 + 8 + i] : (_Float16)0.0f;
    return a;
}

__launch_bounds__(256)
__global__ void involution_fused(const float* __restrict__ x,
                                 const float* __restrict__ reduce_w,
                                 const float* __restrict__ sig_gamma,
                                 const float* __restrict__ sig_beta,
                                 const float* __restrict__ sig_mean,
                                 const float* __restrict__ sig_var,
                                 const float* __restrict__ span_w,
                                 const float* __restrict__ bn_gamma,
                                 const float* __restrict__ bn_beta,
                                 const float* __restrict__ bn_mean,
                                 const float* __restrict__ bn_var,
                                 const float* __restrict__ prelu_a,
                                 float* __restrict__ out) {
    __shared__ float    xs[Cn * CPS];      // x tile + halo, f32      (102,656 B)
    __shared__ _Float16 zs[Cn * ZSTR];     // BN+ReLU'd z, f16        ( 33,024 B)
    __shared__ float    kbuf[KK * 256];    // per-pixel kernels       (  9,216 B)
    __shared__ float    sc1[Cn], sh1[Cn], sc2[Cn], sh2[Cn], pra[Cn];

    const int tid  = threadIdx.x;
    const int lane = tid & 31;
    const int wv   = tid >> 5;
    const int bb   = blockIdx.z;
    const int h0   = blockIdx.y * TS;
    const int w0   = blockIdx.x * TS;
    const float* xb = x + (bb * Cn) * (Hn * Wn);

    // fold BN params (inference form)
    if (tid < Cn) {
        float s1 = sig_gamma[tid] * rsqrtf(sig_var[tid] + 1e-5f);
        sc1[tid] = s1;
        sh1[tid] = sig_beta[tid] - sig_mean[tid] * s1;
        float s2 = bn_gamma[tid] * rsqrtf(bn_var[tid] + 1e-3f);
        sc2[tid] = s2;
        sh2[tid] = bn_beta[tid] - bn_mean[tid] * s2;
        pra[tid] = prelu_a[tid];
    }

    // ---- stage 0: load x tile (+ dilation halo, zero padded) into LDS ----
    for (int idx = tid; idx < Cn * XT * XT; idx += 256) {
        int c  = idx / (XT * XT);
        int q  = idx - c * (XT * XT);
        int ry = q / XT;
        int rx = q - ry * XT;
        int h  = h0 - 2 + ry;
        int w  = w0 - 2 + rx;
        float v = 0.0f;
        if ((unsigned)h < (unsigned)Hn && (unsigned)w < (unsigned)Wn)
            v = xb[c * (Hn * Wn) + h * Wn + w];
        xs[c * CPS + q] = v;
    }
    __syncthreads();

    const int ln15 = lane & 15;
    const int kHiB = (lane >> 4) << 4;     // B-frag: lanes 16-31 hold K 16..31
    const int rHi  = (lane >> 4) << 3;     // C/D: lanes 16-31 hold M = r+8

    // ---- stage 1: z = reduce_w * x  (WMMA), BN+ReLU, f16 -> LDS ----
    v16h A1[4][2];
#pragma unroll
    for (int m = 0; m < 4; ++m)
#pragma unroll
        for (int kb = 0; kb < 2; ++kb)
            A1[m][kb] = load_a_frag(reduce_w, Cn, m * 16, kb * 32, lane, Cn);

#pragma unroll
    for (int t = 0; t < 2; ++t) {
        int nb   = wv * 2 + t;                       // N-block == tile row
        int hidx = (nb + 2) * XT + ln15 + 2;         // pixel in halo coords
        int pix  = nb * 16 + ln15;
        v16h Bf[2];
#pragma unroll
        for (int kb = 0; kb < 2; ++kb) {
            int k0 = kb * 32 + kHiB;
#pragma unroll
            for (int i = 0; i < 16; ++i)
                Bf[kb][i] = (_Float16)xs[(k0 + i) * CPS + hidx];
        }
#pragma unroll
        for (int m = 0; m < 4; ++m) {
            v8f acc = {};
            acc = wmma16(A1[m][0], Bf[0], acc);
            acc = wmma16(A1[m][1], Bf[1], acc);
#pragma unroll
            for (int r = 0; r < 8; ++r) {
                int ch  = m * 16 + r + rHi;
                float v = fmaf(acc[r], sc1[ch], sh1[ch]);
                v = fmaxf(v, 0.0f);
                zs[ch * ZSTR + pix] = (_Float16)v;
            }
        }
    }
    __syncthreads();

    // ---- stage 2: kern = span_w * z  (WMMA, M padded 9->16) ----
    v16h A2[2];
    A2[0] = load_a_frag(span_w, Cn, 0, 0,  lane, KK);
    A2[1] = load_a_frag(span_w, Cn, 0, 32, lane, KK);
#pragma unroll
    for (int t = 0; t < 2; ++t) {
        int nb  = wv * 2 + t;
        int pix = nb * 16 + ln15;
        v16h Bf[2];
#pragma unroll
        for (int kb = 0; kb < 2; ++kb) {
            int k0 = kb * 32 + kHiB;
#pragma unroll
            for (int i = 0; i < 16; ++i)
                Bf[kb][i] = zs[(k0 + i) * ZSTR + pix];
        }
        v8f acc = {};
        acc = wmma16(A2[0], Bf[0], acc);
        acc = wmma16(A2[1], Bf[1], acc);
#pragma unroll
        for (int r = 0; r < 8; ++r) {
            int k = r + rHi;
            if (k < KK) kbuf[k * 256 + pix] = acc[r];
        }
    }
    __syncthreads();

    // ---- stage 3: 9-tap dilated gather + BN + PReLU, write out ----
    {
        int p  = tid;          // one pixel per thread
        int r  = p >> 4;
        int cx = p & 15;
        float kv[KK];
        int   soff[KK];
#pragma unroll
        for (int kh = 0; kh < 3; ++kh)
#pragma unroll
            for (int kw = 0; kw < 3; ++kw) {
                int k   = kh * 3 + kw;
                kv[k]   = kbuf[k * 256 + p];
                soff[k] = (r + 2 * kh) * XT + (cx + 2 * kw);
            }
        float* op = out + (bb * Cn) * (Hn * Wn) + (h0 + r) * Wn + (w0 + cx);
        for (int c = 0; c < Cn; ++c) {
            const float* xc = xs + c * CPS;
            float y = 0.0f;
#pragma unroll
            for (int k = 0; k < KK; ++k)
                y = fmaf(kv[k], xc[soff[k]], y);
            y = fmaf(y, sc2[c], sh2[c]);
            op[c * (Hn * Wn)] = (y >= 0.0f) ? y : pra[c] * y;
        }
    }
}

extern "C" void kernel_launch(void* const* d_in, const int* in_sizes, int n_in,
                              void* d_out, int out_size, void* d_ws, size_t ws_size,
                              hipStream_t stream) {
    const float* x         = (const float*)d_in[0];
    const float* reduce_w  = (const float*)d_in[1];
    const float* sig_gamma = (const float*)d_in[2];
    const float* sig_beta  = (const float*)d_in[3];
    const float* sig_mean  = (const float*)d_in[4];
    const float* sig_var   = (const float*)d_in[5];
    const float* span_w    = (const float*)d_in[6];
    const float* bn_gamma  = (const float*)d_in[7];
    const float* bn_beta   = (const float*)d_in[8];
    const float* bn_mean   = (const float*)d_in[9];
    const float* bn_var    = (const float*)d_in[10];
    const float* prelu_a   = (const float*)d_in[11];
    float* out = (float*)d_out;

    dim3 grid(Wn / TS, Hn / TS, Bn);
    involution_fused<<<grid, 256, 0, stream>>>(
        x, reduce_w, sig_gamma, sig_beta, sig_mean, sig_var, span_w,
        bn_gamma, bn_beta, bn_mean, bn_var, prelu_a, out);
}
